// MoE_Projection_33002528703126
// MI455X (gfx1250) — matrix operand order repaired and checked by reading, not compiled, not run on hardware
//
#include <hip/hip_runtime.h>
#include <hip/hip_bf16.h>

// Sizes fixed by the reference
#define BB 8
#define NN 2048
#define DD 512
#define HH 1024
#define EE 8
#define CAP 512
#define NTOK (BB * NN)

typedef __attribute__((ext_vector_type(16))) __bf16 v16bf;
typedef __attribute__((ext_vector_type(8))) float v8f;

union FragU {
    uint4 q[2];
    v16bf v;
};

// ---- CDNA5 async global->LDS path (guarded: falls back to VGPR copies) ----
#if defined(__has_builtin)
#if __has_builtin(__builtin_amdgcn_global_load_async_to_lds_b128)
#define HAVE_ASYNC 1
#endif
#endif
#ifndef HAVE_ASYNC
#define HAVE_ASYNC 0
#endif

__device__ __forceinline__ void stage16(const unsigned short* g, unsigned short* l) {
#if HAVE_ASYNC
    // param0: int4 addrspace(1)* (global src), param1: int4 addrspace(3)* (LDS dst)
    // generic LDS pointer: low 32 bits are the DS byte offset (ISA 10.2 aperture rule)
    typedef int v4i_t __attribute__((vector_size(16)));
    __builtin_amdgcn_global_load_async_to_lds_b128(
        (__attribute__((address_space(1))) v4i_t*)(unsigned long long)g,
        (__attribute__((address_space(3))) v4i_t*)(unsigned)(unsigned long long)l,
        0, 0);
#else
    *(uint4*)l = *(const uint4*)g;
#endif
}

__device__ __forceinline__ void wait_async_zero() {
#if HAVE_ASYNC
#if __has_builtin(__builtin_amdgcn_s_wait_asynccnt)
    __builtin_amdgcn_s_wait_asynccnt(0);
#else
    asm volatile("s_wait_asynccnt 0x0" ::: "memory");
#endif
#endif
}

__device__ __forceinline__ unsigned short f2bf(float f) {
    unsigned u = __float_as_uint(f);
    u += 0x7FFFu + ((u >> 16) & 1u);   // round-to-nearest-even
    return (unsigned short)(u >> 16);
}

__device__ __forceinline__ unsigned long long pack4(float4 v) {
    return (unsigned long long)f2bf(v.x)
         | ((unsigned long long)f2bf(v.y) << 16)
         | ((unsigned long long)f2bf(v.z) << 32)
         | ((unsigned long long)f2bf(v.w) << 48);
}

// ---------------------------------------------------------------- zero fill
__global__ __launch_bounds__(256) void zeroK(float* p, long long n) {
    long long i = (long long)blockIdx.x * 256 + threadIdx.x;
    long long stride = (long long)gridDim.x * 256;
    for (; i < n; i += stride) p[i] = 0.0f;
}

// ---------------------------------------------------------------- gating: one wave per token
__global__ __launch_bounds__(256) void gating(
    const float* __restrict__ x, const float* __restrict__ probs,
    const float* __restrict__ wg,
    int* __restrict__ idx1, int* __restrict__ idx2,
    float* __restrict__ g1o, float* __restrict__ g2o,
    int* __restrict__ keep2, float* __restrict__ raw8,
    unsigned* __restrict__ cnt1)
{
    __shared__ float swg[DD * EE];   // 16 KB
    int tid = threadIdx.x;
    for (int i = tid; i < DD * EE; i += 256) swg[i] = wg[i];
    __syncthreads();

    int wave = tid >> 5, lane = tid & 31;
    int T = blockIdx.x * 8 + wave;               // token index, grid = NTOK/8

    float acc[EE];
#pragma unroll
    for (int e = 0; e < EE; ++e) acc[e] = 0.0f;

    const float* xr = x + (size_t)T * DD;
#pragma unroll 4
    for (int j = 0; j < DD / 32; ++j) {
        int d = j * 32 + lane;                   // coalesced
        float xv = xr[d];
        const float* wr = &swg[d * EE];
#pragma unroll
        for (int e = 0; e < EE; ++e) acc[e] += xv * wr[e];
    }
#pragma unroll
    for (int e = 0; e < EE; ++e) {
        acc[e] += __shfl_xor(acc[e], 16, 32);
        acc[e] += __shfl_xor(acc[e], 8, 32);
        acc[e] += __shfl_xor(acc[e], 4, 32);
        acc[e] += __shfl_xor(acc[e], 2, 32);
        acc[e] += __shfl_xor(acc[e], 1, 32);
    }
    if (lane == 0) {
        float mx = acc[0];
#pragma unroll
        for (int e = 1; e < EE; ++e) mx = fmaxf(mx, acc[e]);
        float ex[EE], s = 0.0f;
#pragma unroll
        for (int e = 0; e < EE; ++e) { ex[e] = expf(acc[e] - mx); s += ex[e]; }
        float raw[EE];
        float inv = 1.0f / s;
#pragma unroll
        for (int e = 0; e < EE; ++e) raw[e] = ex[e] * inv;

        int i1 = 0; float gm1 = raw[0];
#pragma unroll
        for (int e = 1; e < EE; ++e) if (raw[e] > gm1) { gm1 = raw[e]; i1 = e; }
        int i2 = -1; float gm2 = -1.0f;
#pragma unroll
        for (int e = 0; e < EE; ++e) if (e != i1 && raw[e] > gm2) { gm2 = raw[e]; i2 = e; }

        float den = gm1 + gm2 + 1e-9f;
        float gn1 = gm1 / den, gn2 = gm2 / den;
        float p = probs[T];
        int k2 = (p < gn2 / 0.2f) ? 1 : 0;

        idx1[T] = i1; idx2[T] = i2;
        g1o[T] = gn1; g2o[T] = gn2; keep2[T] = k2;
#pragma unroll
        for (int e = 0; e < EE; ++e) raw8[(size_t)T * EE + e] = raw[e];
        int b = T / NN;
        atomicAdd(&cnt1[b * EE + i1], 1u);
    }
}

// ---------------------------------------------------------------- capacity scan: 1 wave per (e,b)
__global__ void scan(const int* __restrict__ idx1, const int* __restrict__ idx2,
                     const float* __restrict__ g1, const float* __restrict__ g2,
                     const int* __restrict__ keep2,
                     int* __restrict__ slotTok, float* __restrict__ slotGate)
{
    int p = blockIdx.x;               // p = e*8 + b
    int e = p >> 3, b = p & 7;
    int lane = threadIdx.x;
    for (int c = lane; c < CAP; c += 32) {
        slotTok[p * CAP + c] = -1;
        slotGate[p * CAP + c] = 0.0f;
    }
    unsigned lmask = (1u << lane) - 1u;

    unsigned c1 = 0;
    for (int i0 = 0; i0 < NN; i0 += 32) {
        int t = i0 + lane;
        int gi = b * NN + t;
        bool m = (idx1[gi] == e);
        unsigned bal = (unsigned)__ballot(m);
        if (m) {
            unsigned pos = c1 + (unsigned)__popc(bal & lmask);   // exclusive cumsum
            if (pos < CAP) { slotTok[p * CAP + pos] = t; slotGate[p * CAP + pos] = g1[gi]; }
        }
        c1 += (unsigned)__popc(bal);
    }
    unsigned c2 = c1 < (unsigned)CAP ? c1 : (unsigned)CAP;       // m1_count (capped)
    for (int i0 = 0; i0 < NN; i0 += 32) {
        int t = i0 + lane;
        int gi = b * NN + t;
        bool m = keep2[gi] && (idx2[gi] == e);
        unsigned bal = (unsigned)__ballot(m);
        if (m) {
            unsigned pos = c2 + (unsigned)__popc(bal & lmask);
            if (pos < CAP) { slotTok[p * CAP + pos] = t; slotGate[p * CAP + pos] = g2[gi]; }
        }
        c2 += (unsigned)__popc(bal);
    }
}

// ---------------------------------------------------------------- f32 -> bf16 transpose (weights)
__global__ __launch_bounds__(256) void transpose_cvt(
    const float* __restrict__ src, unsigned short* __restrict__ dst, int R, int C)
{
    __shared__ float tile[32][33];
    int mat = blockIdx.z;
    src += (size_t)mat * R * C;
    dst += (size_t)mat * R * C;
    int c0 = blockIdx.x * 32, r0 = blockIdx.y * 32;
    int lx = threadIdx.x & 31, ly = threadIdx.x >> 5;
#pragma unroll
    for (int j = 0; j < 4; ++j)
        tile[ly + j * 8][lx] = src[(size_t)(r0 + ly + j * 8) * C + c0 + lx];
    __syncthreads();
#pragma unroll
    for (int j = 0; j < 4; ++j)
        dst[(size_t)(c0 + ly + j * 8) * R + r0 + lx] = f2bf(tile[lx][ly + j * 8]);
}

#define LDW 40   // padded LDS row stride (elements) for 32-wide K tiles

// ---------------------------------------------------------------- GEMM1: EI(gathered x) x W1 -> LeakyReLU -> hid (bf16)
__global__ __launch_bounds__(256) void gemm1(
    const float* __restrict__ x, const int* __restrict__ slotTok,
    const unsigned short* __restrict__ w1t, unsigned short* __restrict__ hid)
{
    const int p = blockIdx.y, e = p >> 3, b = p & 7;
    const int tm = blockIdx.x & 3, tn = blockIdx.x >> 2;    // 4 M-tiles x 8 N-tiles
    const int tid = threadIdx.x;

    __shared__ __align__(16) unsigned short sA[2][128 * LDW];
    __shared__ __align__(16) unsigned short sB[2][128 * LDW];
    __shared__ int sTok[128];
    if (tid < 128) sTok[tid] = slotTok[p * CAP + tm * 128 + tid];
    __syncthreads();

    const int lane = tid & 31, wid = tid >> 5;
    const int wm = wid >> 2, wn = wid & 3;                  // 2 x 4 wave grid
    const int lr = lane & 15, lh = lane >> 4;

    const int arow = tid >> 3;            // 0..31
    const int acol = (tid & 7) * 4;       // 0,4,..,28

    v8f acc[4][2] = {};
    float4 va[4];

    // ---- prologue: stage k0 = 0 into buffer 0
    {
#pragma unroll
        for (int r0 = 0; r0 < 4; ++r0) {
            int row = arow + r0 * 32;
            int tok = sTok[row];
            va[r0] = make_float4(0.f, 0.f, 0.f, 0.f);
            if (tok >= 0)
                va[r0] = *(const float4*)(x + (size_t)(b * NN + tok) * DD + acol);
        }
#pragma unroll
        for (int r0 = 0; r0 < 4; ++r0) {
            int row = arow + r0 * 32;
            *(unsigned long long*)&sA[0][row * LDW + acol] = pack4(va[r0]);
        }
#pragma unroll
        for (int it = 0; it < 2; ++it) {
            int c = tid + it * 256;
            int row = c >> 2, part = c & 3;
            stage16(w1t + (size_t)(e * HH + tn * 128 + row) * DD + part * 8,
                    &sB[0][row * LDW + part * 8]);
        }
    }

    int buf = 0;
    for (int kk = 0; kk < DD / 32; ++kk) {
        wait_async_zero();
        __syncthreads();
        const int k0n = (kk + 1) * 32;
        const bool more = (kk + 1) < (DD / 32);
        if (more) {
            // issue next-iter global loads early (latency hides under WMMA below)
#pragma unroll
            for (int r0 = 0; r0 < 4; ++r0) {
                int row = arow + r0 * 32;
                int tok = sTok[row];
                va[r0] = make_float4(0.f, 0.f, 0.f, 0.f);
                if (tok >= 0)
                    va[r0] = *(const float4*)(x + (size_t)(b * NN + tok) * DD + k0n + acol);
            }
#pragma unroll
            for (int it = 0; it < 2; ++it) {
                int c = tid + it * 256;
                int row = c >> 2, part = c & 3;
                stage16(w1t + (size_t)(e * HH + tn * 128 + row) * DD + k0n + part * 8,
                        &sB[buf ^ 1][row * LDW + part * 8]);
            }
        }

        FragU af[4], bfm[2];
#pragma unroll
        for (int f = 0; f < 4; ++f) {   // A: lane holds K {0..7,16..23} / {8..15,24..31}
            int base = (wm * 64 + f * 16 + lr) * LDW + lh * 8;
            af[f].q[0] = *(const uint4*)&sA[buf][base];
            af[f].q[1] = *(const uint4*)&sA[buf][base + 16];
        }
#pragma unroll
        for (int g = 0; g < 2; ++g) {   // B: lane holds K {0..7,16..23} / {8..15,24..31}
            int base = (wn * 32 + g * 16 + lr) * LDW + lh * 8;
            bfm[g].q[0] = *(const uint4*)&sB[buf][base];
            bfm[g].q[1] = *(const uint4*)&sB[buf][base + 16];
        }
#pragma unroll
        for (int f = 0; f < 4; ++f)
#pragma unroll
            for (int g = 0; g < 2; ++g)
                acc[f][g] = __builtin_amdgcn_wmma_f32_16x16x32_bf16(
                    false, af[f].v, false, bfm[g].v, (short)0, acc[f][g], false, false);

        if (more) {
#pragma unroll
            for (int r0 = 0; r0 < 4; ++r0) {
                int row = arow + r0 * 32;
                *(unsigned long long*)&sA[buf ^ 1][row * LDW + acol] = pack4(va[r0]);
            }
        }
        buf ^= 1;
    }
    // epilogue: LeakyReLU + cvt to bf16
#pragma unroll
    for (int f = 0; f < 4; ++f)
#pragma unroll
        for (int g = 0; g < 2; ++g)
#pragma unroll
            for (int vi = 0; vi < 8; ++vi) {
                int m = tm * 128 + wm * 64 + f * 16 + lh * 8 + vi;
                int n = tn * 128 + wn * 32 + g * 16 + lr;
                float val = acc[f][g][vi];
                val = val >= 0.0f ? val : 0.01f * val;
                hid[(size_t)(p * CAP + m) * HH + n] = f2bf(val);
            }
}

// ---------------------------------------------------------------- GEMM2: hid x W2 -> scatter-add out with gates
__global__ __launch_bounds__(256) void gemm2(
    const unsigned short* __restrict__ hid, const int* __restrict__ slotTok,
    const float* __restrict__ slotGate, const unsigned short* __restrict__ w2t,
    float* __restrict__ out)
{
    const int p = blockIdx.y, e = p >> 3, b = p & 7;
    const int tm = blockIdx.x & 3, tn = blockIdx.x >> 2;    // 4 x 4 tiles
    const int tid = threadIdx.x;

    __shared__ __align__(16) unsigned short sA[2][128 * LDW];
    __shared__ __align__(16) unsigned short sB[2][128 * LDW];
    __shared__ int sTok[128];
    __shared__ float sGate[128];
    if (tid < 128) {
        sTok[tid] = slotTok[p * CAP + tm * 128 + tid];
        sGate[tid] = slotGate[p * CAP + tm * 128 + tid];
    }

    const int lane = tid & 31, wid = tid >> 5;
    const int wm = wid >> 2, wn = wid & 3;
    const int lr = lane & 15, lh = lane >> 4;

    v8f acc[4][2] = {};

    // ---- prologue: stage k0 = 0 into buffer 0 (both tiles are pure copies)
#pragma unroll
    for (int it = 0; it < 2; ++it) {
        int c = tid + it * 256;
        int row = c >> 2, part = c & 3;
        stage16(hid + (size_t)(p * CAP + tm * 128 + row) * HH + part * 8,
                &sA[0][row * LDW + part * 8]);
        stage16(w2t + (size_t)(e * DD + tn * 128 + row) * HH + part * 8,
                &sB[0][row * LDW + part * 8]);
    }

    int buf = 0;
    for (int kk = 0; kk < HH / 32; ++kk) {
        wait_async_zero();
        __syncthreads();
        const int k0n = (kk + 1) * 32;
        if ((kk + 1) < (HH / 32)) {
#pragma unroll
            for (int it = 0; it < 2; ++it) {
                int c = tid + it * 256;
                int row = c >> 2, part = c & 3;
                stage16(hid + (size_t)(p * CAP + tm * 128 + row) * HH + k0n + part * 8,
                        &sA[buf ^ 1][row * LDW + part * 8]);
                stage16(w2t + (size_t)(e * DD + tn * 128 + row) * HH + k0n + part * 8,
                        &sB[buf ^ 1][row * LDW + part * 8]);
            }
        }

        FragU af[4], bfm[2];
#pragma unroll
        for (int f = 0; f < 4; ++f) {
            int base = (wm * 64 + f * 16 + lr) * LDW + lh * 8;
            af[f].q[0] = *(const uint4*)&sA[buf][base];
            af[f].q[1] = *(const uint4*)&sA[buf][base + 16];
        }
#pragma unroll
        for (int g = 0; g < 2; ++g) {
            int base = (wn * 32 + g * 16 + lr) * LDW + lh * 8;
            bfm[g].q[0] = *(const uint4*)&sB[buf][base];
            bfm[g].q[1] = *(const uint4*)&sB[buf][base + 16];
        }
#pragma unroll
        for (int f = 0; f < 4; ++f)
#pragma unroll
            for (int g = 0; g < 2; ++g)
                acc[f][g] = __builtin_amdgcn_wmma_f32_16x16x32_bf16(
                    false, af[f].v, false, bfm[g].v, (short)0, acc[f][g], false, false);
        buf ^= 1;
    }
    // epilogue: out[b, tok, dcol] += gate * val   (<=2 adds/elem -> commutative -> deterministic)
#pragma unroll
    for (int f = 0; f < 4; ++f)
#pragma unroll
        for (int g = 0; g < 2; ++g)
#pragma unroll
            for (int vi = 0; vi < 8; ++vi) {
                int lm = wm * 64 + f * 16 + lh * 8 + vi;    // local row in [0,128)
                int t = sTok[lm];
                if (t >= 0) {
                    int dcol = tn * 128 + wn * 32 + g * 16 + lr;
                    unsafeAtomicAdd(out + (size_t)(b * NN + t) * DD + dcol,
                                    sGate[lm] * acc[f][g][vi]);
                }
            }
}

// ---------------------------------------------------------------- aux loss (deterministic reductions)
__global__ __launch_bounds__(256) void loss_part(
    const float* __restrict__ raw8, const unsigned* __restrict__ cnt1, float* __restrict__ terms)
{
    int p = blockIdx.x;              // p = b*8 + e
    int b = p >> 3, e = p & 7;
    __shared__ float red[256];
    float s = 0.0f;
    for (int n = threadIdx.x; n < NN; n += 256)
        s += raw8[(size_t)(b * NN + n) * EE + e];
    red[threadIdx.x] = s;
    __syncthreads();
    for (int st = 128; st > 0; st >>= 1) {
        if ((int)threadIdx.x < st) red[threadIdx.x] += red[threadIdx.x + st];
        __syncthreads();
    }
    if (threadIdx.x == 0)
        terms[p] = (red[0] * (1.0f / NN)) * ((float)cnt1[p] * (1.0f / NN));
}

__global__ void loss_final(const float* __restrict__ terms, float* __restrict__ lossOut)
{
    __shared__ float red[64];
    int t = threadIdx.x;
    red[t] = terms[t];
    __syncthreads();
    for (int s = 32; s > 0; s >>= 1) {
        if (t < s) red[t] += red[t + s];
        __syncthreads();
    }
    // mean over 64 * e*e(=64) * LOSS_COEF(0.01) == sum * 0.01
    if (t == 0) *lossOut = red[0] * 0.01f;
}

// ---------------------------------------------------------------- launcher
extern "C" void kernel_launch(void* const* d_in, const int* in_sizes, int n_in,
                              void* d_out, int out_size, void* d_ws, size_t ws_size,
                              hipStream_t stream)
{
    const float* x     = (const float*)d_in[0];
    const float* probs = (const float*)d_in[1];
    const float* wg    = (const float*)d_in[2];
    const float* w1    = (const float*)d_in[3];
    const float* w2    = (const float*)d_in[4];
    float* out = (float*)d_out;

    char* base = (char*)d_ws;
    size_t off = 0;
    auto alloc = [&](size_t bytes) -> char* {
        char* r = base + off;
        off = (off + bytes + 255) & ~(size_t)255;
        return r;
    };
    unsigned* cnt1        = (unsigned*)alloc(64 * 4);
    float* terms          = (float*)alloc(64 * 4);
    int* idx1             = (int*)alloc((size_t)NTOK * 4);
    int* idx2             = (int*)alloc((size_t)NTOK * 4);
    float* g1             = (float*)alloc((size_t)NTOK * 4);
    float* g2             = (float*)alloc((size_t)NTOK * 4);
    int* keep2            = (int*)alloc((size_t)NTOK * 4);
    float* raw8           = (float*)alloc((size_t)NTOK * EE * 4);
    int* slotTok          = (int*)alloc((size_t)EE * BB * CAP * 4);
    float* slotGate       = (float*)alloc((size_t)EE * BB * CAP * 4);
    unsigned short* w1t   = (unsigned short*)alloc((size_t)EE * HH * DD * 2);
    unsigned short* w2t   = (unsigned short*)alloc((size_t)EE * DD * HH * 2);
    unsigned short* hid   = (unsigned short*)alloc((size_t)EE * BB * CAP * HH * 2);

    zeroK<<<2048, 256, 0, stream>>>(out, (long long)out_size);
    zeroK<<<1, 256, 0, stream>>>((float*)cnt1, 64);

    gating<<<NTOK / 8, 256, 0, stream>>>(x, probs, wg, idx1, idx2, g1, g2, keep2, raw8, cnt1);
    scan<<<EE * BB, 32, 0, stream>>>(idx1, idx2, g1, g2, keep2, slotTok, slotGate);

    transpose_cvt<<<dim3(HH / 32, DD / 32, EE), 256, 0, stream>>>(w1, w1t, DD, HH); // w1t[e][h][d]
    transpose_cvt<<<dim3(DD / 32, HH / 32, EE), 256, 0, stream>>>(w2, w2t, HH, DD); // w2t[e][d][h]

    gemm1<<<dim3(32, EE * BB), 256, 0, stream>>>(x, slotTok, w1t, hid);
    gemm2<<<dim3(16, EE * BB), 256, 0, stream>>>(hid, slotTok, slotGate, w2t, out);

    loss_part<<<BB * EE, 256, 0, stream>>>(raw8, cnt1, terms);
    loss_final<<<1, 64, 0, stream>>>(terms, out + (size_t)(out_size - 1));
}
